// LinearAttention_43224550868161
// MI455X (gfx1250) — matrix operand order repaired and checked by reading, not compiled, not run on hardware
//
#include <hip/hip_runtime.h>
#include <hip/hip_bf16.h>

// ---------------------------------------------------------------------------
// Linear attention (shared-state) for MI455X / gfx1250, wave32 + WMMA bf16.
//
// Pipeline (all on `stream`, stream order provides dependencies):
//   1. cvt      : x, Wq, Wk, Wv  f32 -> bf16 workspace
//   2. qkv_gemm : Q = elu(x Wq^T)+1, K = elu(x Wk^T)+1, V = x Wv^T   (bf16)
//   3. denom    : denom[t,b] = sum_d Q[t,b,d] * cumsum_t K[t,b,d]
//   4. chunk_kv : G_c = K_c^T V_c per chunk (C = 512 flattened positions)
//   5. prefix   : in-place exclusive prefix over chunks -> S_c (state)
//   6. scores   : P_c = blockcausal_mask(Q_c K_c^T)                   (bf16)
//   7. out      : y = (Q_c S_c + P_c V_c) / denom                     (f32)
//
// GEMM kernels: 256 threads (8 wave32), 128x128 tile, wave owns 32x64
// (2x4 WMMA 16x16x32 bf16 tiles). K staged 32-wide through double-buffered
// LDS panels; panel fill uses GLOBAL_LOAD_ASYNC_TO_LDS_B128 when available.
// ---------------------------------------------------------------------------

typedef __bf16 bf16;
typedef __bf16  bf16x16 __attribute__((ext_vector_type(16)));
typedef float   f32x8   __attribute__((ext_vector_type(8)));
typedef int     v4i     __attribute__((vector_size(16)));

#define T_DIM   4096
#define B_DIM   4
#define D       1024
#define NT      (T_DIM * B_DIM)   // 16384 flattened positions
#define CHUNK   512
#define NCHUNK  (NT / CHUNK)      // 32

__device__ __forceinline__ float bf2f(bf16 x) { return (float)x; }
__device__ __forceinline__ bf16  f2bf(float x) { return (bf16)x; }

union FragBF { bf16x16 v; uint4 q[2]; };

// ---- CDNA5 async global->LDS path (guarded; sync fallback) -----------------
#if __has_builtin(__builtin_amdgcn_global_load_async_to_lds_b128)
#define HAVE_ASYNC_LDS 1
#endif

// copy 16 bytes global -> LDS (per-lane); async unit writes LDS directly
__device__ __forceinline__ void async_copy16(const bf16* g, bf16* lds) {
#ifdef HAVE_ASYNC_LDS
  __builtin_amdgcn_global_load_async_to_lds_b128(
      (__attribute__((address_space(1))) v4i*)(g),
      (__attribute__((address_space(3))) v4i*)(lds),
      0, 0);
#else
  *(uint4*)lds = *(const uint4*)g;
#endif
}

__device__ __forceinline__ void wait_async0() {
#ifdef HAVE_ASYNC_LDS
#if __has_builtin(__builtin_amdgcn_s_wait_asynccnt)
  __builtin_amdgcn_s_wait_asynccnt(0);
#else
  asm volatile("s_wait_asynccnt 0x0" ::: "memory");
#endif
#endif
}

// ---- LDS staging: panels A[128][32], BT[128][32] (row stride 32 bf16) ------

// 128 rows x 32 cols, source rows contiguous (row stride gstride elems)
__device__ __forceinline__ void stage_rows_async(bf16* lds, const bf16* g,
                                                 int gstride, int tid) {
  int r = tid >> 1;
  int h = (tid & 1) * 16;
  const bf16* src = g + (size_t)r * gstride + h;
  bf16* dst = lds + r * 32 + h;
  async_copy16(src,     dst);
  async_copy16(src + 8, dst + 8);
}

// source block is [32 k][128 n] row-major; write LDS transposed as [n][k]
// (transpose must pass through VGPRs, so this path stays synchronous)
__device__ __forceinline__ void stage_T(bf16* lds, const bf16* g,
                                        int gstride, int tid) {
  int k  = tid >> 3;           // 0..31
  int n0 = (tid & 7) * 16;     // 0..112
  const bf16* src = g + (size_t)k * gstride + n0;
  uint4 v0 = *(const uint4*)(src);
  uint4 v1 = *(const uint4*)(src + 8);
  bf16 tmp[16];
  *(uint4*)(tmp)     = v0;
  *(uint4*)(tmp + 8) = v1;
#pragma unroll
  for (int j = 0; j < 16; ++j) lds[(n0 + j) * 32 + k] = tmp[j];
}

// ---- WMMA fragment loads (ISA 7.12.2 layouts) ------------------------------
__device__ __forceinline__ FragBF load_fragA(const bf16* ldsA, int mrow, int lane) {
  FragBF f;
  int m = mrow + (lane & 15);
  int h = (lane >> 4) * 8;
  const bf16* p = ldsA + m * 32;
  f.q[0] = *(const uint4*)(p + h);
  f.q[1] = *(const uint4*)(p + 16 + h);
  return f;
}
__device__ __forceinline__ FragBF load_fragB(const bf16* ldsBT, int ncol, int lane) {
  FragBF f;
  int n = ncol + (lane & 15);
  int g = (lane >> 4) * 8;
  const bf16* p = ldsBT + n * 32 + g;
  f.q[0] = *(const uint4*)(p);
  f.q[1] = *(const uint4*)(p + 16);
  return f;
}

// one K=32 step: wave computes its 32x64 sub-tile (2x4 WMMA tiles)
__device__ __forceinline__ void wmma_step(const bf16* ldsA, const bf16* ldsBT,
                                          int mbase, int nbase, int lane,
                                          f32x8 (&acc)[2][4]) {
  FragBF a[2], b[4];
#pragma unroll
  for (int mt = 0; mt < 2; ++mt) a[mt] = load_fragA(ldsA, mbase + mt * 16, lane);
#pragma unroll
  for (int nt = 0; nt < 4; ++nt) b[nt] = load_fragB(ldsBT, nbase + nt * 16, lane);
#pragma unroll
  for (int mt = 0; mt < 2; ++mt)
#pragma unroll
    for (int nt = 0; nt < 4; ++nt)
      acc[mt][nt] = __builtin_amdgcn_wmma_f32_16x16x32_bf16(
          false, a[mt].v, false, b[nt].v, (short)0, acc[mt][nt], false, false);
}

__device__ __forceinline__ void zero_acc(f32x8 (&acc)[2][4]) {
#pragma unroll
  for (int i = 0; i < 2; ++i)
#pragma unroll
    for (int j = 0; j < 4; ++j)
#pragma unroll
      for (int r = 0; r < 8; ++r) acc[i][j][r] = 0.0f;
}

// ---------------------------------------------------------------------------
// 1. f32 -> bf16 conversion
__global__ void cvt_f32_bf16(const float* __restrict__ in, bf16* __restrict__ out,
                             int n) {
  int i = blockIdx.x * blockDim.x + threadIdx.x;
  if (i < n) out[i] = f2bf(in[i]);
}

// ---------------------------------------------------------------------------
// 2. QKV projection: Out[n,o] = f(sum_d x[n,d] * W[o,d]); elu+1 for Q,K
//    Both panels row-natural -> fully async staged, double buffered.
__global__ __launch_bounds__(256) void qkv_gemm(
    const bf16* __restrict__ xb, const bf16* __restrict__ wq,
    const bf16* __restrict__ wk, const bf16* __restrict__ wv,
    bf16* __restrict__ Qb, bf16* __restrict__ Kb, bf16* __restrict__ Vb) {
  __shared__ __align__(16) bf16 ldsA[2][128 * 32];
  __shared__ __align__(16) bf16 ldsBT[2][128 * 32];
  int tid = threadIdx.x, lane = tid & 31, w = tid >> 5;
  int mbase = (w >> 1) * 32, nbase = (w & 1) * 64;
  int rowBlk = blockIdx.x * 128;   // over NT
  int colBlk = blockIdx.y * 128;   // over D (output features)
  int which  = blockIdx.z;
  const bf16* W = (which == 0) ? wq : (which == 1) ? wk : wv;
  bf16* Out     = (which == 0) ? Qb : (which == 1) ? Kb : Vb;
  const bf16* Arow = xb + (size_t)rowBlk * D;
  const bf16* Brow = W  + (size_t)colBlk * D;

  f32x8 acc[2][4];
  zero_acc(acc);
  const int NK = D / 32;
  stage_rows_async(ldsA[0],  Arow, D, tid);
  stage_rows_async(ldsBT[0], Brow, D, tid);
  wait_async0();
  __syncthreads();
  for (int i = 0; i < NK; ++i) {
    int cur = i & 1;
    if (i + 1 < NK) {
      int k0 = (i + 1) * 32;
      stage_rows_async(ldsA[cur ^ 1],  Arow + k0, D, tid);
      stage_rows_async(ldsBT[cur ^ 1], Brow + k0, D, tid);
    }
    wmma_step(ldsA[cur], ldsBT[cur], mbase, nbase, lane, acc);
    if (i + 1 < NK) {
      wait_async0();
      __syncthreads();
    }
  }
  bool act = (which < 2);
#pragma unroll
  for (int mt = 0; mt < 2; ++mt)
#pragma unroll
    for (int nt = 0; nt < 4; ++nt)
#pragma unroll
      for (int r = 0; r < 8; ++r) {
        float v = acc[mt][nt][r];
        if (act) v = (v > 0.0f) ? (v + 1.0f) : __expf(v);  // elu(v)+1
        int row = rowBlk + mbase + mt * 16 + r + 8 * (lane >> 4);
        int col = colBlk + nbase + nt * 16 + (lane & 15);
        Out[(size_t)row * D + col] = f2bf(v);
      }
}

// ---------------------------------------------------------------------------
// 3a. denom partials: per (b, slice of 64 d), scan T keeping z in registers
__global__ void denom_partial(const bf16* __restrict__ Qb,
                              const bf16* __restrict__ Kb,
                              float* __restrict__ partial) {
  int b = blockIdx.x, slice = blockIdx.y;
  int lane = threadIdx.x;            // 32 threads, each owns 2 d-indices
  int d0 = slice * 64 + lane;
  float z0 = 0.0f, z1 = 0.0f;
#pragma unroll 4
  for (int t = 0; t < T_DIM; ++t) {
    size_t base = ((size_t)(t * B_DIM + b)) * D;
    // prefetch 16 timesteps ahead (stays inside the bf16 workspace)
    __builtin_prefetch(Kb + base + (size_t)16 * B_DIM * D + d0, 0, 0);
    __builtin_prefetch(Qb + base + (size_t)16 * B_DIM * D + d0, 0, 0);
    z0 += bf2f(Kb[base + d0]);
    z1 += bf2f(Kb[base + d0 + 32]);
    float part = bf2f(Qb[base + d0]) * z0 + bf2f(Qb[base + d0 + 32]) * z1;
#pragma unroll
    for (int off = 16; off > 0; off >>= 1) part += __shfl_xor(part, off);
    if (lane == 0) partial[((size_t)(t * B_DIM + b)) * 16 + slice] = part;
  }
}

// 3b. reduce 16 slices -> denom[p]
__global__ void denom_reduce(const float* __restrict__ partial,
                             float* __restrict__ denom) {
  int i = blockIdx.x * blockDim.x + threadIdx.x;
  if (i < NT) {
    float s = 0.0f;
#pragma unroll
    for (int j = 0; j < 16; ++j) s += partial[(size_t)i * 16 + j];
    denom[i] = s;
  }
}

// ---------------------------------------------------------------------------
// 4. per-chunk G_c[dk,dm] = sum_{p in chunk} K[p,dk] V[p,dm]
//    (both panels transposed -> synchronous staging)
__global__ __launch_bounds__(256) void chunk_kv(const bf16* __restrict__ Kb,
                                                const bf16* __restrict__ Vb,
                                                bf16* __restrict__ G) {
  __shared__ __align__(16) bf16 ldsA[128 * 32];   // [dk][p]
  __shared__ __align__(16) bf16 ldsBT[128 * 32];  // [dm][p]
  int tid = threadIdx.x, lane = tid & 31, w = tid >> 5;
  int mbase = (w >> 1) * 32, nbase = (w & 1) * 64;
  int c = blockIdx.z;
  int dk0 = blockIdx.x * 128, dm0 = blockIdx.y * 128;

  f32x8 acc[2][4];
  zero_acc(acc);
  for (int p0 = 0; p0 < CHUNK; p0 += 32) {
    __syncthreads();
    const bf16* ksrc = Kb + (size_t)(c * CHUNK + p0) * D + dk0;
    const bf16* vsrc = Vb + (size_t)(c * CHUNK + p0) * D + dm0;
    stage_T(ldsA,  ksrc, D, tid);
    stage_T(ldsBT, vsrc, D, tid);
    __syncthreads();
    wmma_step(ldsA, ldsBT, mbase, nbase, lane, acc);
  }
#pragma unroll
  for (int mt = 0; mt < 2; ++mt)
#pragma unroll
    for (int nt = 0; nt < 4; ++nt)
#pragma unroll
      for (int r = 0; r < 8; ++r) {
        int dk = dk0 + mbase + mt * 16 + r + 8 * (lane >> 4);
        int dm = dm0 + nbase + nt * 16 + (lane & 15);
        G[((size_t)c * D + dk) * D + dm] = f2bf(acc[mt][nt][r]);
      }
}

// ---------------------------------------------------------------------------
// 5. in-place exclusive prefix over chunks (f32 running sum, bf16 storage)
__global__ void prefix_chunks(bf16* __restrict__ G) {
  int e = blockIdx.x * blockDim.x + threadIdx.x;  // element within [D*D]
  float run = 0.0f;
  for (int c = 0; c < NCHUNK; ++c) {
    size_t idx = (size_t)c * D * D + e;
    float v = bf2f(G[idx]);
    G[idx] = f2bf(run);
    run += v;
  }
}

// ---------------------------------------------------------------------------
// 6. intra-chunk scores: P[pq,pk] = (t(pq) >= t(pk)) ? Q[pq]·K[pk] : 0
//    Both panels row-natural -> fully async staged, double buffered.
__global__ __launch_bounds__(256) void chunk_scores(const bf16* __restrict__ Qb,
                                                    const bf16* __restrict__ Kb,
                                                    bf16* __restrict__ P) {
  __shared__ __align__(16) bf16 ldsA[2][128 * 32];
  __shared__ __align__(16) bf16 ldsBT[2][128 * 32];
  int tid = threadIdx.x, lane = tid & 31, w = tid >> 5;
  int mbase = (w >> 1) * 32, nbase = (w & 1) * 64;
  int c = blockIdx.z;
  int pq0 = blockIdx.x * 128, pk0 = blockIdx.y * 128;  // local in chunk
  const bf16* Arow = Qb + (size_t)(c * CHUNK + pq0) * D;
  const bf16* Brow = Kb + (size_t)(c * CHUNK + pk0) * D;

  f32x8 acc[2][4];
  zero_acc(acc);
  const int NK = D / 32;
  stage_rows_async(ldsA[0],  Arow, D, tid);
  stage_rows_async(ldsBT[0], Brow, D, tid);
  wait_async0();
  __syncthreads();
  for (int i = 0; i < NK; ++i) {
    int cur = i & 1;
    if (i + 1 < NK) {
      int k0 = (i + 1) * 32;
      stage_rows_async(ldsA[cur ^ 1],  Arow + k0, D, tid);
      stage_rows_async(ldsBT[cur ^ 1], Brow + k0, D, tid);
    }
    wmma_step(ldsA[cur], ldsBT[cur], mbase, nbase, lane, acc);
    if (i + 1 < NK) {
      wait_async0();
      __syncthreads();
    }
  }
#pragma unroll
  for (int mt = 0; mt < 2; ++mt)
#pragma unroll
    for (int nt = 0; nt < 4; ++nt)
#pragma unroll
      for (int r = 0; r < 8; ++r) {
        int ql = pq0 + mbase + mt * 16 + r + 8 * (lane >> 4);
        int kl = pk0 + nbase + nt * 16 + (lane & 15);
        // t = p / B_DIM; same-chunk comparison
        float v = ((ql >> 2) >= (kl >> 2)) ? acc[mt][nt][r] : 0.0f;
        P[(size_t)c * CHUNK * CHUNK + (size_t)ql * CHUNK + kl] = f2bf(v);
      }
}

// ---------------------------------------------------------------------------
// 7. Out = (Q_c @ S_c + P_c @ V_c) / denom
//    A panel async (row-natural), B panel transposed (sync); double buffered.
__global__ __launch_bounds__(256) void chunk_out(
    const bf16* __restrict__ Qb, const bf16* __restrict__ Vb,
    const bf16* __restrict__ P, const bf16* __restrict__ S,
    const float* __restrict__ denom, float* __restrict__ out) {
  __shared__ __align__(16) bf16 ldsA[2][128 * 32];
  __shared__ __align__(16) bf16 ldsBT[2][128 * 32];
  int tid = threadIdx.x, lane = tid & 31, w = tid >> 5;
  int mbase = (w >> 1) * 32, nbase = (w & 1) * 64;
  int c = blockIdx.z;
  int pq0 = blockIdx.x * 128;   // local row block within chunk
  int dm0 = blockIdx.y * 128;   // output feature block

  const int NK1 = D / 32;       // phase 1: Q_c @ S_c
  const int NK2 = CHUNK / 32;   // phase 2: P_c @ V_c
  const int NK  = NK1 + NK2;

  auto stage = [&](int i, int buf) {
    if (i < NK1) {
      int k0 = i * 32;
      stage_rows_async(ldsA[buf], Qb + (size_t)(c * CHUNK + pq0) * D + k0, D, tid);
      stage_T(ldsBT[buf], S + (size_t)c * D * D + (size_t)k0 * D + dm0, D, tid);
    } else {
      int k0 = (i - NK1) * 32;
      stage_rows_async(ldsA[buf],
                       P + (size_t)c * CHUNK * CHUNK + (size_t)pq0 * CHUNK + k0,
                       CHUNK, tid);
      stage_T(ldsBT[buf], Vb + (size_t)(c * CHUNK + k0) * D + dm0, D, tid);
    }
  };

  f32x8 acc[2][4];
  zero_acc(acc);
  stage(0, 0);
  wait_async0();
  __syncthreads();
  for (int i = 0; i < NK; ++i) {
    int cur = i & 1;
    if (i + 1 < NK) stage(i + 1, cur ^ 1);
    wmma_step(ldsA[cur], ldsBT[cur], mbase, nbase, lane, acc);
    if (i + 1 < NK) {
      wait_async0();
      __syncthreads();
    }
  }
#pragma unroll
  for (int mt = 0; mt < 2; ++mt)
#pragma unroll
    for (int nt = 0; nt < 4; ++nt)
#pragma unroll
      for (int r = 0; r < 8; ++r) {
        int ql = pq0 + mbase + mt * 16 + r + 8 * (lane >> 4);
        int p  = c * CHUNK + ql;
        int dm = dm0 + nbase + nt * 16 + (lane & 15);
        out[(size_t)p * D + dm] = acc[mt][nt][r] / denom[p];
      }
}

// ---------------------------------------------------------------------------
extern "C" void kernel_launch(void* const* d_in, const int* in_sizes, int n_in,
                              void* d_out, int out_size, void* d_ws,
                              size_t ws_size, hipStream_t stream) {
  (void)in_sizes; (void)n_in; (void)out_size; (void)ws_size;
  const float* x  = (const float*)d_in[0];
  const float* Wq = (const float*)d_in[1];
  const float* Wk = (const float*)d_in[2];
  const float* Wv = (const float*)d_in[3];
  float* out = (float*)d_out;

  // workspace layout (bytes); total ~215 MiB
  char* ws = (char*)d_ws;
  bf16*  xb      = (bf16*)(ws);                          // 32 MiB
  bf16*  wqb     = (bf16*)(ws + 33554432);               //  2 MiB
  bf16*  wkb     = (bf16*)(ws + 35651584);               //  2 MiB
  bf16*  wvb     = (bf16*)(ws + 37748736);               //  2 MiB
  bf16*  Qb      = (bf16*)(ws + 39845888);               // 32 MiB
  bf16*  Kb      = (bf16*)(ws + 73400320);               // 32 MiB
  bf16*  Vb      = (bf16*)(ws + 106954752);              // 32 MiB
  bf16*  G       = (bf16*)(ws + 140509184);              // 64 MiB (states)
  bf16*  Pm      = (bf16*)(ws + 207618048);              // 16 MiB (scores)
  float* partial = (float*)(ws + 224395264);             //  1 MiB
  float* denom   = (float*)(ws + 225443840);             // 64 KiB

  // 1. convert to bf16
  cvt_f32_bf16<<<(NT * D + 255) / 256, 256, 0, stream>>>(x, xb, NT * D);
  cvt_f32_bf16<<<(D * D + 255) / 256, 256, 0, stream>>>(Wq, wqb, D * D);
  cvt_f32_bf16<<<(D * D + 255) / 256, 256, 0, stream>>>(Wk, wkb, D * D);
  cvt_f32_bf16<<<(D * D + 255) / 256, 256, 0, stream>>>(Wv, wvb, D * D);

  // 2. Q/K/V projections (z selects which)
  qkv_gemm<<<dim3(NT / 128, D / 128, 3), 256, 0, stream>>>(xb, wqb, wkb, wvb,
                                                           Qb, Kb, Vb);
  // 3. denominator
  denom_partial<<<dim3(B_DIM, 16), 32, 0, stream>>>(Qb, Kb, partial);
  denom_reduce<<<(NT + 255) / 256, 256, 0, stream>>>(partial, denom);

  // 4. per-chunk K^T V
  chunk_kv<<<dim3(D / 128, D / 128, NCHUNK), 256, 0, stream>>>(Kb, Vb, G);

  // 5. exclusive prefix over chunks (in place)
  prefix_chunks<<<(D * D) / 256, 256, 0, stream>>>(G);

  // 6. intra-chunk masked scores
  chunk_scores<<<dim3(CHUNK / 128, CHUNK / 128, NCHUNK), 256, 0, stream>>>(Qb, Kb,
                                                                          Pm);
  // 7. output
  chunk_out<<<dim3(CHUNK / 128, D / 128, NCHUNK), 256, 0, stream>>>(
      Qb, Vb, Pm, G, denom, out);
}
